// GraphGradModel_9277129359617
// MI455X (gfx1250) — matrix operand order repaired and checked by reading, not compiled, not run on hardware
//
#include <hip/hip_runtime.h>
#include <hip/hip_bf16.h>

typedef __attribute__((ext_vector_type(16))) _Float16 v16h;
typedef __attribute__((ext_vector_type(8)))  _Float16 v8h;
typedef __attribute__((ext_vector_type(8)))  float    v8f;

namespace {
constexpr int kN0 = 1024, kN1 = 512, kN2 = 256, kN3 = 128;
constexpr int kF = 64, kFin = 16, kB = 8, kNE = 16384;
}

// ---------------------------------------------------------------------------
// Tiled batched WMMA GEMM: C[MxN] = A[MxK] @ B[KxN], f16 in, f32 out.
// 256 threads = 8 waves per block; 64x64 C tile; K-step 32.
// Tiles staged to LDS with GLOBAL_LOAD_ASYNC_TO_LDS_B128 (ASYNCcnt),
// double-buffered: next stage issued before waiting on the current one
// (async ops complete in order, so s_wait_asynccnt <= 2 releases stage k
// while stage k+1 is still in flight).
// Requires M,N multiples of 64 and K multiple of 32.
// LDS layout (dynamic, 16 KB): buf{0,1}: [A 64x32 f16 (4KB) | B 32x64 f16 (4KB)]
// ---------------------------------------------------------------------------
__global__ void __launch_bounds__(256)
wmma_gemm_tiled(const _Float16* __restrict__ A, const _Float16* __restrict__ B,
                float* __restrict__ C, int K, int lda, int ldb, int ldc,
                long long sA, long long sB, long long sC) {
  extern __shared__ char smem[];
  const int tid = threadIdx.x;
  const int b = blockIdx.z;
  const int m0 = blockIdx.y << 6;
  const int n0 = blockIdx.x << 6;
  const _Float16* Ag = A + (long long)b * sA;
  const _Float16* Bg = B + (long long)b * sB;

  // staging coordinates: one 16-byte async copy per thread per matrix
  const int rowA = tid >> 2, segA = (tid & 3) << 3;   // 64 rows x 4 x 8 halfs
  const int rowB = tid >> 3, segB = (tid & 7) << 3;   // 32 rows x 8 x 8 halfs
  const unsigned ldsA = (unsigned)(rowA * 32 + segA) * 2u;           // bytes
  const unsigned ldsB = 4096u + (unsigned)(rowB * 64 + segB) * 2u;   // bytes

  const int lane = tid & 31;
  const int wave = tid >> 5;
  const int tr = wave >> 1;          // wave's 16-row tile (0..3)
  const int tcb = (wave & 1) << 1;   // wave's first 16-col tile (0 or 2)
  const int lrow = lane & 15;
  const int kA = (lane >> 4) << 3;   // A frag k-base: 0 or 8
  const int kB2 = (lane >> 4) << 3;  // B frag k-base: 0 or 8

  v8f acc0 = {0, 0, 0, 0, 0, 0, 0, 0};
  v8f acc1 = {0, 0, 0, 0, 0, 0, 0, 0};

  const int nk = K >> 5;
  {  // issue stage 0 into buffer 0
    const _Float16* ga = Ag + (long long)(m0 + rowA) * lda + segA;
    const _Float16* gb = Bg + (long long)rowB * ldb + n0 + segB;
    asm volatile("global_load_async_to_lds_b128 %0, %1, off" ::"v"(ldsA), "v"(ga) : "memory");
    asm volatile("global_load_async_to_lds_b128 %0, %1, off" ::"v"(ldsB), "v"(gb) : "memory");
  }
  for (int ks = 0; ks < nk; ++ks) {
    const unsigned base = (ks & 1) ? 8192u : 0u;
    if (ks + 1 < nk) {  // issue next stage into the other buffer
      const unsigned nb = ((ks + 1) & 1) ? 8192u : 0u;
      const int k0 = (ks + 1) << 5;
      const _Float16* ga = Ag + (long long)(m0 + rowA) * lda + k0 + segA;
      const _Float16* gb = Bg + (long long)(k0 + rowB) * ldb + n0 + segB;
      asm volatile("global_load_async_to_lds_b128 %0, %1, off" ::"v"(ldsA + nb), "v"(ga) : "memory");
      asm volatile("global_load_async_to_lds_b128 %0, %1, off" ::"v"(ldsB + nb), "v"(gb) : "memory");
      asm volatile("s_wait_asynccnt 0x2" ::: "memory");  // stage ks done; next in flight
    } else {
      asm volatile("s_wait_asynccnt 0x0" ::: "memory");
    }
    __syncthreads();

    const _Float16* sAt = (const _Float16*)(smem + base);
    const _Float16* sBt = (const _Float16*)(smem + base + 4096);

    // A fragment (ISA 16-bit A 16x32 layout), shared by both column tiles
    const int ar = (tr << 4) + lrow;
    v8h alo = *(const v8h*)(sAt + ar * 32 + kA);
    v8h ahi = *(const v8h*)(sAt + ar * 32 + kA + 16);
    v16h af = __builtin_shufflevector(alo, ahi, 0, 1, 2, 3, 4, 5, 6, 7,
                                      8, 9, 10, 11, 12, 13, 14, 15);
    // B fragments (ISA 16-bit B 32x16 layout) for the two column tiles
    v16h bf0, bf1;
#pragma unroll
    for (int e = 0; e < 16; ++e) {
      bf0[e] = sBt[(kB2 + (e & 7) + ((e >> 3) << 4)) * 64 + (tcb << 4) + lrow];
      bf1[e] = sBt[(kB2 + (e & 7) + ((e >> 3) << 4)) * 64 + ((tcb + 1) << 4) + lrow];
    }
    acc0 = __builtin_amdgcn_wmma_f32_16x16x32_f16(false, af, false, bf0, (short)0, acc0, false, false);
    acc1 = __builtin_amdgcn_wmma_f32_16x16x32_f16(false, af, false, bf1, (short)0, acc1, false, false);
    __syncthreads();
  }
  const int crow = m0 + (tr << 4) + ((lane >> 4) << 3);
  float* Cp0 = C + (long long)b * sC + (long long)crow * ldc + n0 + (tcb << 4) + lrow;
  float* Cp1 = Cp0 + 16;
#pragma unroll
  for (int r = 0; r < 8; ++r) {
    Cp0[(long long)r * ldc] = acc0[r];
    Cp1[(long long)r * ldc] = acc1[r];
  }
}

// ---------------------------------------------------------------------------
// Simple one-wave WMMA GEMM for small-N cases (N==16 output layers).
// ---------------------------------------------------------------------------
__global__ void wmma_gemm_f16(const _Float16* __restrict__ A,
                              const _Float16* __restrict__ B,
                              float* __restrict__ C,
                              int K, int lda, int ldb, int ldc,
                              long long sA, long long sB, long long sC) {
  const int lane = threadIdx.x;
  const int b = blockIdx.z;
  const int row = (blockIdx.y << 4) + (lane & 15);
  const int col = (blockIdx.x << 4) + (lane & 15);
  const int kA = (lane >> 4) << 3;
  const int kB = (lane >> 4) << 3;

  const _Float16* Ap = A + (long long)b * sA + (long long)row * lda + kA;
  const _Float16* Bp = B + (long long)b * sB + (long long)kB * ldb + col;

  v8f acc = {0.f, 0.f, 0.f, 0.f, 0.f, 0.f, 0.f, 0.f};
  for (int k0 = 0; k0 < K; k0 += 32) {
    v16h af, bf;
#pragma unroll
    for (int j = 0; j < 8; ++j) af[j] = Ap[j];
#pragma unroll
    for (int j = 0; j < 8; ++j) af[8 + j] = Ap[16 + j];
#pragma unroll
    for (int e = 0; e < 16; ++e) bf[e] = Bp[(long long)((e & 7) + ((e >> 3) << 4)) * ldb];
    acc = __builtin_amdgcn_wmma_f32_16x16x32_f16(false, af, false, bf,
                                                 (short)0, acc, false, false);
    Ap += 32;
    Bp += (long long)32 * ldb;
  }
  const int crow = (blockIdx.y << 4) + ((lane >> 4) << 3);
  float* Cp = C + (long long)b * sC + (long long)crow * ldc + col;
#pragma unroll
  for (int r = 0; r < 8; ++r) Cp[(long long)r * ldc] = acc[r];
}

// ---------------------------------------------------------------------------
// Support kernels
// ---------------------------------------------------------------------------
__global__ void sumsq_kernel(const float* __restrict__ x, float* stats, int n) {
  __shared__ float sm[256];
  int tid = threadIdx.x;
  float s = 0.f;
  for (int i = blockIdx.x * blockDim.x + tid; i < n; i += gridDim.x * blockDim.x) {
    float v = x[i];
    s += v * v;
  }
  sm[tid] = s;
  __syncthreads();
  for (int w = 128; w > 0; w >>= 1) {
    if (tid < w) sm[tid] += sm[tid + w];
    __syncthreads();
  }
  if (tid == 0) atomicAdd(&stats[0], sm[0]);
}

__global__ void finalize_norm(float* stats, float inv_count) {
  if (threadIdx.x == 0 && blockIdx.x == 0)
    stats[1] = rsqrtf(stats[0] * inv_count);  // 1 / sqrt(mean(x^2))
}

__global__ void scatter_edges(const int* __restrict__ e, float* __restrict__ A,
                              int ne, int n) {
  int k = blockIdx.x * blockDim.x + threadIdx.x;
  if (k >= ne) return;
  int s = e[k], d = e[ne + k];
  atomicAdd(&A[(long long)s * n + d], 1.0f);
}

// pad/convert f32 -> f16 (zero pad), optional scalar scale read from device
__global__ void stage_pad_f16(const float* __restrict__ in, _Float16* __restrict__ out,
                              int rowsIn, int colsIn, int rowsOut, int colsOut,
                              long long sIn, long long sOut,
                              const float* __restrict__ scale) {
  int b = blockIdx.y;
  int idx = blockIdx.x * blockDim.x + threadIdx.x;
  int total = rowsOut * colsOut;
  if (idx >= total) return;
  int r = idx / colsOut, c = idx - r * colsOut;
  float v = 0.f;
  if (r < rowsIn && c < colsIn) {
    v = in[(long long)b * sIn + (long long)r * colsIn + c];
    if (scale) v *= scale[0];
  }
  out[(long long)b * sOut + idx] = (_Float16)v;
}

// di[i] = rsqrt(2 + rowsum(A))   (A has zero diagonal; +2I self loops)
__global__ void degree_rsqrt(const float* __restrict__ A, float* __restrict__ di,
                             int n, long long sA) {
  int b = blockIdx.y;
  int i = blockIdx.x * blockDim.x + threadIdx.x;
  if (i >= n) return;
  const float* row = A + (long long)b * sA + (long long)i * n;
  float d = 2.0f;
  for (int j = 0; j < n; ++j) d += row[j];
  di[b * n + i] = rsqrtf(d);
}

// An_h[i][j] = f16( di[i] * (A[i][j] + 2*(i==j)) * di[j] )
__global__ void normalize_adj_f16(const float* __restrict__ A, const float* __restrict__ di,
                                  _Float16* __restrict__ out, int n,
                                  long long sA, long long sO) {
  int b = blockIdx.y;
  long long idx = (long long)blockIdx.x * blockDim.x + threadIdx.x;
  if (idx >= (long long)n * n) return;
  int i = (int)(idx / n), j = (int)(idx - (long long)i * n);
  float v = A[(long long)b * sA + idx] + (i == j ? 2.0f : 0.0f);
  out[(long long)b * sO + idx] = (_Float16)(di[b * n + i] * v * di[b * n + j]);
}

// B_h = f16( A with diagonal forced to 1 )   (augment: B = A*(1-I)+I)
__global__ void stage_aug_f16(const float* __restrict__ A, _Float16* __restrict__ out,
                              int n, long long sA, long long sO) {
  int b = blockIdx.y;
  long long idx = (long long)blockIdx.x * blockDim.x + threadIdx.x;
  if (idx >= (long long)n * n) return;
  int i = (int)(idx / n), j = (int)(idx - (long long)i * n);
  float v = (i == j) ? 1.0f : A[(long long)b * sA + idx];
  out[(long long)b * sO + idx] = (_Float16)v;
}

__global__ void zero_diag(float* __restrict__ A, int n, long long sA) {
  int b = blockIdx.y;
  int i = blockIdx.x * blockDim.x + threadIdx.x;
  if (i < n) A[(long long)b * sA + (long long)i * n + i] = 0.f;
}

// score = tanh( (x @ w) / ||w|| )
__global__ void pool_score(const float* __restrict__ x, const float* __restrict__ w,
                           float* __restrict__ score, int n) {
  int b = blockIdx.y;
  int i = blockIdx.x * blockDim.x + threadIdx.x;
  if (i >= n) return;
  float wn = 0.f;
  for (int f = 0; f < kF; ++f) wn += w[f] * w[f];
  wn = sqrtf(wn);
  const float* xr = x + ((long long)b * n + i) * kF;
  float s = 0.f;
  for (int f = 0; f < kF; ++f) s += xr[f] * w[f];
  score[b * n + i] = tanhf(s / wn);
}

// bitonic sort per batch: descending score, ascending index ties (jax top_k order)
__global__ void topk_kernel(const float* __restrict__ score,
                            float* __restrict__ topv, int* __restrict__ perm,
                            int n, int k) {
  __shared__ float skey[1024];
  __shared__ int sidx[1024];
  int b = blockIdx.x;
  int t = threadIdx.x;
  skey[t] = score[b * n + t];
  sidx[t] = t;
  __syncthreads();
  for (int size = 2; size <= n; size <<= 1) {
    for (int stride = size >> 1; stride > 0; stride >>= 1) {
      int j = t ^ stride;
      if (j > t) {
        float ka = skey[t], kb = skey[j];
        int ia = sidx[t], ib = sidx[j];
        bool aBefore = (ka > kb) || (ka == kb && ia < ib);
        bool up = ((t & size) == 0);
        if (up ? !aBefore : aBefore) {
          skey[t] = kb; sidx[t] = ib;
          skey[j] = ka; sidx[j] = ia;
        }
      }
      __syncthreads();
    }
  }
  if (t < k) {
    topv[b * k + t] = skey[t];
    perm[b * k + t] = sidx[t];
  }
}

__global__ void pool_gather_x(const float* __restrict__ xin, const int* __restrict__ perm,
                              const float* __restrict__ topv, float* __restrict__ xout,
                              int nin, int k) {
  int b = blockIdx.y;
  int idx = blockIdx.x * blockDim.x + threadIdx.x;
  if (idx >= k * kF) return;
  int i = idx / kF, f = idx - i * kF;
  int p = perm[b * k + i];
  xout[((long long)b * k + i) * kF + f] =
      xin[((long long)b * nin + p) * kF + f] * topv[b * k + i];
}

__global__ void pool_gather_A(const float* __restrict__ Ain, const int* __restrict__ perm,
                              float* __restrict__ Aout, int nin, int nout, long long sIn) {
  int b = blockIdx.y;
  int idx = blockIdx.x * blockDim.x + threadIdx.x;
  if (idx >= nout * nout) return;
  int i = idx / nout, j = idx - i * nout;
  int pi = perm[b * nout + i], pj = perm[b * nout + j];
  Aout[(long long)b * nout * nout + idx] =
      Ain[(long long)b * sIn + (long long)pi * nin + pj];
}

__global__ void bias_act(float* __restrict__ y, const float* __restrict__ bias,
                         long long total, int F, int relu) {
  long long idx = (long long)blockIdx.x * blockDim.x + threadIdx.x;
  if (idx >= total) return;
  float v = y[idx] + bias[(int)(idx % F)];
  y[idx] = (relu && v < 0.f) ? 0.f : v;
}

__global__ void unpool_scatter_add(float* __restrict__ u, const float* __restrict__ xs,
                                   const int* __restrict__ perm, int k, int nbig) {
  int b = blockIdx.y;
  int idx = blockIdx.x * blockDim.x + threadIdx.x;
  if (idx >= k * kF) return;
  int i = idx / kF, f = idx - i * kF;
  u[((long long)b * nbig + perm[b * k + i]) * kF + f] +=
      xs[((long long)b * k + i) * kF + f];
}

// ---------------------------------------------------------------------------
// Host orchestration
// ---------------------------------------------------------------------------
extern "C" void kernel_launch(void* const* d_in, const int* in_sizes, int n_in,
                              void* d_out, int out_size, void* d_ws, size_t ws_size,
                              hipStream_t stream) {
  (void)in_sizes; (void)n_in; (void)out_size; (void)ws_size;
  const float* x    = (const float*)d_in[0];
  const int*   edge = (const int*)d_in[1];
  const float* Wd0 = (const float*)d_in[2];  const float* bd0 = (const float*)d_in[3];
  const float* Wd1 = (const float*)d_in[4];  const float* bd1 = (const float*)d_in[5];
  const float* Wd2 = (const float*)d_in[6];  const float* bd2 = (const float*)d_in[7];
  const float* Wd3 = (const float*)d_in[8];  const float* bd3 = (const float*)d_in[9];
  const float* pw1 = (const float*)d_in[10];
  const float* pw2 = (const float*)d_in[11];
  const float* pw3 = (const float*)d_in[12];
  const float* Wu0 = (const float*)d_in[13]; const float* bu0 = (const float*)d_in[14];
  const float* Wu1 = (const float*)d_in[15]; const float* bu1 = (const float*)d_in[16];
  const float* Wu2 = (const float*)d_in[17]; const float* bu2 = (const float*)d_in[18];
  float* out = (float*)d_out;

  // --- bump allocator over workspace ---
  char* ws = (char*)d_ws;
  size_t off = 0;
  auto alloc = [&](size_t bytes) -> void* {
    void* p = ws + off;
    off = (off + bytes + 255) & ~(size_t)255;
    return p;
  };
  float*    stats = (float*)alloc(2 * sizeof(float));
  float*    di    = (float*)alloc((size_t)kB * kN0 * 4);
  float*    A0    = (float*)alloc((size_t)kN0 * kN0 * 4);
  _Float16* An0h  = (_Float16*)alloc((size_t)kN0 * kN0 * 2);
  float*    A1    = (float*)alloc((size_t)kB * kN1 * kN1 * 4);
  _Float16* An1h  = (_Float16*)alloc((size_t)kB * kN1 * kN1 * 2);
  float*    A2    = (float*)alloc((size_t)kB * kN2 * kN2 * 4);
  _Float16* An2h  = (_Float16*)alloc((size_t)kB * kN2 * kN2 * 2);
  float*    A3    = (float*)alloc((size_t)kB * kN3 * kN3 * 4);
  _Float16* An3h  = (_Float16*)alloc((size_t)kB * kN3 * kN3 * 2);
  float*    Aaug  = (float*)alloc((size_t)kB * kN1 * kN1 * 4);     // reused per level
  _Float16* Bh    = (_Float16*)alloc((size_t)kB * kN1 * kN1 * 2);  // reused
  float*    score = (float*)alloc((size_t)kB * kN0 * 4);
  int*      perm1 = (int*)alloc((size_t)kB * kN1 * 4);
  float*    topv1 = (float*)alloc((size_t)kB * kN1 * 4);
  int*      perm2 = (int*)alloc((size_t)kB * kN2 * 4);
  float*    topv2 = (float*)alloc((size_t)kB * kN2 * 4);
  int*      perm3 = (int*)alloc((size_t)kB * kN3 * 4);
  float*    topv3 = (float*)alloc((size_t)kB * kN3 * 4);
  float*    x0 = (float*)alloc((size_t)kB * kN0 * kF * 4);
  float*    x1 = (float*)alloc((size_t)kB * kN1 * kF * 4);
  float*    x2 = (float*)alloc((size_t)kB * kN2 * kF * 4);
  float*    x3 = (float*)alloc((size_t)kB * kN3 * kF * 4);
  float*    xp = (float*)alloc((size_t)kB * kN1 * kF * 4);       // pooled feats (reused)
  float*    t  = (float*)alloc((size_t)kB * kN0 * kF * 4);       // xW intermediate
  float*    u  = (float*)alloc((size_t)kB * kN0 * kF * 4);       // unpooled feats
  float*    g  = (float*)alloc((size_t)kB * kN1 * kF * 4);       // up-GCN output
  _Float16* fh = (_Float16*)alloc((size_t)kB * kN0 * kF * 2);    // staged GEMM A
  _Float16* th = (_Float16*)alloc((size_t)kB * kN0 * kF * 2);    // staged GEMM B
  _Float16* xnh = (_Float16*)alloc((size_t)kB * kN0 * 32 * 2);   // x/||.|| padded 16->32
  _Float16* wd0h = (_Float16*)alloc(32 * kF * 2);
  _Float16* wd1h = (_Float16*)alloc(kF * kF * 2);
  _Float16* wd2h = (_Float16*)alloc(kF * kF * 2);
  _Float16* wd3h = (_Float16*)alloc(kF * kF * 2);
  _Float16* wu0h = (_Float16*)alloc(kF * kF * 2);
  _Float16* wu1h = (_Float16*)alloc(kF * kF * 2);
  _Float16* wu2h = (_Float16*)alloc(kF * kFin * 2);

  auto gemm = [&](const _Float16* Am, const _Float16* Bm, float* Cm,
                  int M, int N, int K, int lda, int ldb, int ldc,
                  long long stA, long long stB, long long stC, int batch) {
    if ((M & 63) == 0 && (N & 63) == 0) {
      dim3 grid(N / 64, M / 64, batch);
      wmma_gemm_tiled<<<grid, dim3(256), 16384, stream>>>(Am, Bm, Cm, K, lda, ldb, ldc,
                                                          stA, stB, stC);
    } else {
      dim3 grid(N / 16, M / 16, batch);
      wmma_gemm_f16<<<grid, dim3(32), 0, stream>>>(Am, Bm, Cm, K, lda, ldb, ldc,
                                                   stA, stB, stC);
    }
  };
  auto stage = [&](const float* in, _Float16* o, int ri, int ci, int ro, int co,
                   long long sIn, long long sOut, int batch, const float* scale) {
    dim3 grid((ro * co + 255) / 256, batch);
    stage_pad_f16<<<grid, 256, 0, stream>>>(in, o, ri, ci, ro, co, sIn, sOut, scale);
  };

  // 1) grad_norm = sqrt(mean(x^2)); store 1/grad_norm in stats[1]
  hipMemsetAsync(stats, 0, 2 * sizeof(float), stream);
  sumsq_kernel<<<64, 256, 0, stream>>>(x, stats, kB * kN0 * kFin);
  finalize_norm<<<1, 32, 0, stream>>>(stats, 1.0f / (float)(kB * kN0 * kFin));

  // 2) dense adjacency
  hipMemsetAsync(A0, 0, (size_t)kN0 * kN0 * 4, stream);
  scatter_edges<<<(kNE + 255) / 256, 256, 0, stream>>>(edge, A0, kNE, kN0);

  // 3) An0 (normalized, f16) — used in down-0 and up-2
  degree_rsqrt<<<dim3((kN0 + 255) / 256, 1), 256, 0, stream>>>(A0, di, kN0, 0);
  normalize_adj_f16<<<dim3((kN0 * kN0 + 255) / 256, 1), 256, 0, stream>>>(A0, di, An0h, kN0, 0, 0);

  // 4) stage weights to f16 (pad Wd0 K: 16 -> 32)
  stage(Wd0, wd0h, kFin, kF, 32, kF, 0, 0, 1, nullptr);
  stage(Wd1, wd1h, kF, kF, kF, kF, 0, 0, 1, nullptr);
  stage(Wd2, wd2h, kF, kF, kF, kF, 0, 0, 1, nullptr);
  stage(Wd3, wd3h, kF, kF, kF, kF, 0, 0, 1, nullptr);
  stage(Wu0, wu0h, kF, kF, kF, kF, 0, 0, 1, nullptr);
  stage(Wu1, wu1h, kF, kF, kF, kF, 0, 0, 1, nullptr);
  stage(Wu2, wu2h, kF, kFin, kF, kFin, 0, 0, 1, nullptr);

  // 5) xn = x / grad_norm, staged f16 padded to K=32
  stage(x, xnh, kN0, kFin, kN0, 32, (long long)kN0 * kFin, (long long)kN0 * 32, kB, stats + 1);

  // 6) down-0 GCN: x0 = relu(An0 @ (xn @ Wd0) + bd0)
  gemm(xnh, wd0h, t, kN0, kF, 32, 32, kF, kF, (long long)kN0 * 32, 0, (long long)kN0 * kF, kB);
  stage(t, th, kN0, kF, kN0, kF, (long long)kN0 * kF, (long long)kN0 * kF, kB, nullptr);
  gemm(An0h, th, x0, kN0, kF, kN0, kN0, kF, kF, 0, (long long)kN0 * kF, (long long)kN0 * kF, kB);
  bias_act<<<(kB * kN0 * kF + 255) / 256, 256, 0, stream>>>(x0, bd0, (long long)kB * kN0 * kF, kF, 1);

  // 7) augment level 0 (shared): Aaug = (A0+I)@(A0+I), zero diag
  stage_aug_f16<<<dim3((kN0 * kN0 + 255) / 256, 1), 256, 0, stream>>>(A0, Bh, kN0, 0, 0);
  gemm(Bh, Bh, Aaug, kN0, kN0, kN0, kN0, kN0, kN0, 0, 0, 0, 1);
  zero_diag<<<dim3((kN0 + 255) / 256, 1), 256, 0, stream>>>(Aaug, kN0, 0);

  // 8) pool 1: top-512
  pool_score<<<dim3((kN0 + 255) / 256, kB), 256, 0, stream>>>(x0, pw1, score, kN0);
  topk_kernel<<<kB, kN0, 0, stream>>>(score, topv1, perm1, kN0, kN1);
  pool_gather_x<<<dim3((kN1 * kF + 255) / 256, kB), 256, 0, stream>>>(x0, perm1, topv1, xp, kN0, kN1);
  pool_gather_A<<<dim3((kN1 * kN1 + 255) / 256, kB), 256, 0, stream>>>(Aaug, perm1, A1, kN0, kN1, 0);

  // 9) down-1 GCN on 512 nodes
  degree_rsqrt<<<dim3((kN1 + 255) / 256, kB), 256, 0, stream>>>(A1, di, kN1, (long long)kN1 * kN1);
  normalize_adj_f16<<<dim3((kN1 * kN1 + 255) / 256, kB), 256, 0, stream>>>(
      A1, di, An1h, kN1, (long long)kN1 * kN1, (long long)kN1 * kN1);
  stage(xp, fh, kN1, kF, kN1, kF, (long long)kN1 * kF, (long long)kN1 * kF, kB, nullptr);
  gemm(fh, wd1h, t, kN1, kF, kF, kF, kF, kF, (long long)kN1 * kF, 0, (long long)kN1 * kF, kB);
  stage(t, th, kN1, kF, kN1, kF, (long long)kN1 * kF, (long long)kN1 * kF, kB, nullptr);
  gemm(An1h, th, x1, kN1, kF, kN1, kN1, kF, kF, (long long)kN1 * kN1, (long long)kN1 * kF, (long long)kN1 * kF, kB);
  bias_act<<<(kB * kN1 * kF + 255) / 256, 256, 0, stream>>>(x1, bd1, (long long)kB * kN1 * kF, kF, 1);

  // 10) augment level 1 (per batch): 8 x 512^3
  stage_aug_f16<<<dim3((kN1 * kN1 + 255) / 256, kB), 256, 0, stream>>>(
      A1, Bh, kN1, (long long)kN1 * kN1, (long long)kN1 * kN1);
  gemm(Bh, Bh, Aaug, kN1, kN1, kN1, kN1, kN1, kN1,
       (long long)kN1 * kN1, (long long)kN1 * kN1, (long long)kN1 * kN1, kB);
  zero_diag<<<dim3((kN1 + 255) / 256, kB), 256, 0, stream>>>(Aaug, kN1, (long long)kN1 * kN1);

  // 11) pool 2: top-256
  pool_score<<<dim3((kN1 + 255) / 256, kB), 256, 0, stream>>>(x1, pw2, score, kN1);
  topk_kernel<<<kB, kN1, 0, stream>>>(score, topv2, perm2, kN1, kN2);
  pool_gather_x<<<dim3((kN2 * kF + 255) / 256, kB), 256, 0, stream>>>(x1, perm2, topv2, xp, kN1, kN2);
  pool_gather_A<<<dim3((kN2 * kN2 + 255) / 256, kB), 256, 0, stream>>>(
      Aaug, perm2, A2, kN1, kN2, (long long)kN1 * kN1);

  // 12) down-2 GCN on 256 nodes
  degree_rsqrt<<<dim3((kN2 + 255) / 256, kB), 256, 0, stream>>>(A2, di, kN2, (long long)kN2 * kN2);
  normalize_adj_f16<<<dim3((kN2 * kN2 + 255) / 256, kB), 256, 0, stream>>>(
      A2, di, An2h, kN2, (long long)kN2 * kN2, (long long)kN2 * kN2);
  stage(xp, fh, kN2, kF, kN2, kF, (long long)kN2 * kF, (long long)kN2 * kF, kB, nullptr);
  gemm(fh, wd2h, t, kN2, kF, kF, kF, kF, kF, (long long)kN2 * kF, 0, (long long)kN2 * kF, kB);
  stage(t, th, kN2, kF, kN2, kF, (long long)kN2 * kF, (long long)kN2 * kF, kB, nullptr);
  gemm(An2h, th, x2, kN2, kF, kN2, kN2, kF, kF, (long long)kN2 * kN2, (long long)kN2 * kF, (long long)kN2 * kF, kB);
  bias_act<<<(kB * kN2 * kF + 255) / 256, 256, 0, stream>>>(x2, bd2, (long long)kB * kN2 * kF, kF, 1);

  // 13) augment level 2 (per batch)
  stage_aug_f16<<<dim3((kN2 * kN2 + 255) / 256, kB), 256, 0, stream>>>(
      A2, Bh, kN2, (long long)kN2 * kN2, (long long)kN2 * kN2);
  gemm(Bh, Bh, Aaug, kN2, kN2, kN2, kN2, kN2, kN2,
       (long long)kN2 * kN2, (long long)kN2 * kN2, (long long)kN2 * kN2, kB);
  zero_diag<<<dim3((kN2 + 255) / 256, kB), 256, 0, stream>>>(Aaug, kN2, (long long)kN2 * kN2);

  // 14) pool 3: top-128
  pool_score<<<dim3((kN2 + 255) / 256, kB), 256, 0, stream>>>(x2, pw3, score, kN2);
  topk_kernel<<<kB, kN2, 0, stream>>>(score, topv3, perm3, kN2, kN3);
  pool_gather_x<<<dim3((kN3 * kF + 255) / 256, kB), 256, 0, stream>>>(x2, perm3, topv3, xp, kN2, kN3);
  pool_gather_A<<<dim3((kN3 * kN3 + 255) / 256, kB), 256, 0, stream>>>(
      Aaug, perm3, A3, kN2, kN3, (long long)kN2 * kN2);

  // 15) down-3 GCN on 128 nodes -> x3 (bottleneck)
  degree_rsqrt<<<dim3((kN3 + 255) / 256, kB), 256, 0, stream>>>(A3, di, kN3, (long long)kN3 * kN3);
  normalize_adj_f16<<<dim3((kN3 * kN3 + 255) / 256, kB), 256, 0, stream>>>(
      A3, di, An3h, kN3, (long long)kN3 * kN3, (long long)kN3 * kN3);
  stage(xp, fh, kN3, kF, kN3, kF, (long long)kN3 * kF, (long long)kN3 * kF, kB, nullptr);
  gemm(fh, wd3h, t, kN3, kF, kF, kF, kF, kF, (long long)kN3 * kF, 0, (long long)kN3 * kF, kB);
  stage(t, th, kN3, kF, kN3, kF, (long long)kN3 * kF, (long long)kN3 * kF, kB, nullptr);
  gemm(An3h, th, x3, kN3, kF, kN3, kN3, kF, kF, (long long)kN3 * kN3, (long long)kN3 * kF, (long long)kN3 * kF, kB);
  bias_act<<<(kB * kN3 * kF + 255) / 256, 256, 0, stream>>>(x3, bd3, (long long)kB * kN3 * kF, kF, 1);

  // 16) up-0 (j=2, 256 nodes): u = x2 + unpool(x3); g = relu(An2 @ (u@Wu0) + bu0)
  hipMemcpyAsync(u, x2, (size_t)kB * kN2 * kF * 4, hipMemcpyDeviceToDevice, stream);
  unpool_scatter_add<<<dim3((kN3 * kF + 255) / 256, kB), 256, 0, stream>>>(u, x3, perm3, kN3, kN2);
  stage(u, fh, kN2, kF, kN2, kF, (long long)kN2 * kF, (long long)kN2 * kF, kB, nullptr);
  gemm(fh, wu0h, t, kN2, kF, kF, kF, kF, kF, (long long)kN2 * kF, 0, (long long)kN2 * kF, kB);
  stage(t, th, kN2, kF, kN2, kF, (long long)kN2 * kF, (long long)kN2 * kF, kB, nullptr);
  gemm(An2h, th, g, kN2, kF, kN2, kN2, kF, kF, (long long)kN2 * kN2, (long long)kN2 * kF, (long long)kN2 * kF, kB);
  bias_act<<<(kB * kN2 * kF + 255) / 256, 256, 0, stream>>>(g, bu0, (long long)kB * kN2 * kF, kF, 1);

  // 17) up-1 (j=1, 512 nodes)
  hipMemcpyAsync(u, x1, (size_t)kB * kN1 * kF * 4, hipMemcpyDeviceToDevice, stream);
  unpool_scatter_add<<<dim3((kN2 * kF + 255) / 256, kB), 256, 0, stream>>>(u, g, perm2, kN2, kN1);
  stage(u, fh, kN1, kF, kN1, kF, (long long)kN1 * kF, (long long)kN1 * kF, kB, nullptr);
  gemm(fh, wu1h, t, kN1, kF, kF, kF, kF, kF, (long long)kN1 * kF, 0, (long long)kN1 * kF, kB);
  stage(t, th, kN1, kF, kN1, kF, (long long)kN1 * kF, (long long)kN1 * kF, kB, nullptr);
  gemm(An1h, th, g, kN1, kF, kN1, kN1, kF, kF, (long long)kN1 * kN1, (long long)kN1 * kF, (long long)kN1 * kF, kB);
  bias_act<<<(kB * kN1 * kF + 255) / 256, 256, 0, stream>>>(g, bu1, (long long)kB * kN1 * kF, kF, 1);

  // 18) up-2 (j=0, 1024 nodes, no relu): out = An0 @ (u@Wu2) + bu2
  hipMemcpyAsync(u, x0, (size_t)kB * kN0 * kF * 4, hipMemcpyDeviceToDevice, stream);
  unpool_scatter_add<<<dim3((kN1 * kF + 255) / 256, kB), 256, 0, stream>>>(u, g, perm1, kN1, kN0);
  stage(u, fh, kN0, kF, kN0, kF, (long long)kN0 * kF, (long long)kN0 * kF, kB, nullptr);
  gemm(fh, wu2h, t, kN0, kFin, kF, kF, kFin, kFin, (long long)kN0 * kF, 0, (long long)kN0 * kFin, kB);
  stage(t, th, kN0, kFin, kN0, kFin, (long long)kN0 * kFin, (long long)kN0 * kFin, kB, nullptr);
  gemm(An0h, th, out, kN0, kFin, kN0, kN0, kFin, kFin, 0, (long long)kN0 * kFin, (long long)kN0 * kFin, kB);
  bias_act<<<(kB * kN0 * kFin + 255) / 256, 256, 0, stream>>>(out, bu2, (long long)kB * kN0 * kFin, kFin, 0);
}
